// GCNLSTMLayer_4380866642246
// MI455X (gfx1250) — matrix operand order repaired and checked by reading, not compiled, not run on hardware
//
#include <hip/hip_runtime.h>
#include <cmath>

#define N_NODES 100000
#define D       128        // hidden size
#define NEDGE   600000

typedef __attribute__((ext_vector_type(16))) _Float16 v16h;
typedef __attribute__((ext_vector_type(8)))  float    v8f;

// ---------------------------------------------------------------------------
// A-fragment (16x32 f16, M x K): lane group 0 (lanes 0-15) holds K = 0..7 in
// halves 0..7 and K = 16..23 in halves 8..15; lane group 1 holds K = 8..15 and
// 24..31.  p points at (row, k0); base = (lane>=16)*8.
__device__ __forceinline__ v16h make_a(const float* __restrict__ p, int base) {
  v16h a;
#pragma unroll
  for (int i = 0; i < 8; ++i) a[i] = (_Float16)p[base + i];
#pragma unroll
  for (int i = 0; i < 8; ++i) a[8 + i] = (_Float16)p[base + 16 + i];
  return a;
}

__device__ __forceinline__ float sigmoidf_(float x) {
  return 1.0f / (1.0f + __expf(-x));
}

__device__ __forceinline__ void atomAddF(float* p, float v) {
#if defined(__gfx1250__)
  unsafeAtomicAdd(p, v);   // global_atomic_add_f32
#else
  atomicAdd(p, v);
#endif
}

// ---------------------------------------------------------------------------
__global__ void zero_kernel(float* __restrict__ p, long n) {
  long i = (long)blockIdx.x * blockDim.x + threadIdx.x;
  if (i < n) p[i] = 0.0f;
}

// One wave per edge: 32 lanes x float4 = 128 channels, atomically accumulated
// into msg[dst]; lane 0 also bumps the degree counter.
__global__ void scatter_kernel(const float* __restrict__ feat,
                               const int* __restrict__ src,
                               const int* __restrict__ dst,
                               float* __restrict__ msg,
                               float* __restrict__ deg) {
  long tid = (long)blockIdx.x * blockDim.x + threadIdx.x;
  long e = tid >> 5;
  if (e >= NEDGE) return;
  int lane = (int)(tid & 31);
  int s = src[e], d = dst[e];
  const float4 v = *reinterpret_cast<const float4*>(feat + (long)s * D + lane * 4);
  float* mp = msg + (long)d * D + lane * 4;
  atomAddF(mp + 0, v.x);
  atomAddF(mp + 1, v.y);
  atomAddF(mp + 2, v.z);
  atomAddF(mp + 3, v.w);
  if (lane == 0) atomAddF(deg + d, 1.0f);
}

__global__ void mean_kernel(float* __restrict__ msg, const float* __restrict__ deg) {
  long i = (long)blockIdx.x * blockDim.x + threadIdx.x;
  if (i >= (long)N_NODES * D) return;
  msg[i] = msg[i] / fmaxf(deg[i >> 7], 1.0f);
}

// ---------------------------------------------------------------------------
// Pack a [512, K] f32 weight matrix into f16 WMMA B-fragments, grouped by
// k-step so each k-step's 32 column tiles form one contiguous 32KB block:
//   P[((ks*32 + ct)*32 + lane)*16 + i] =
//       W[ct*16 + (lane&15)][ks*32 + (lane>>4)*8 + (i&7) + (i>>3)*16]
__global__ void pack_w_kernel(const float* __restrict__ W,
                              _Float16* __restrict__ P,
                              int kSteps, long total) {
  long t = (long)blockIdx.x * blockDim.x + threadIdx.x;
  if (t >= total) return;
  int i    = (int)(t & 15);
  int lane = (int)((t >> 4) & 31);
  long grp = t >> 9;                 // ks*32 + ct
  int ct = (int)(grp & 31);
  int ks = (int)(grp >> 5);
  int row = ct * 16 + (lane & 15);
  int col = ks * 32 + (lane >> 4) * 8 + (i & 7) + (i >> 3) * 16;
  P[t] = (_Float16)W[(long)row * (kSteps * 32) + col];
}

// ---------------------------------------------------------------------------
// Fused GEMM + LSTM cell.  KXS is compile-time so every B-fragment load is an
// immediate offset off Wbase + lane*32B.  One wave per 16-row node tile; four
// passes, each holding 2 channel-tiles x 4 gates = 8 f32 accumulator tiles
// (64 VGPRs) so ~2/3 of the register file stays free for the scheduler to
// keep B-fragment loads in flight across WMMAs (staggered s_wait_loadcnt).
template <int KXS, bool HAS_XB, bool DUP>
__global__ __launch_bounds__(128) void lstm_layer_kernel(
    const float* __restrict__ Xa,        // cols 0..127 of ih input
    const float* __restrict__ Xb,        // cols 128..255 (unused if !HAS_XB)
    const float* __restrict__ H0,        // [N,128] h0[l]
    const float* __restrict__ C0,        // [N,128] c0[l]
    const _Float16* __restrict__ Wih,    // packed B-fragments [KXS][32] tiles
    const _Float16* __restrict__ Whh,    // packed B-fragments [4][32] tiles
    const float* __restrict__ Bih,       // [512]
    const float* __restrict__ Bhh,       // [512]
    float* __restrict__ Hout,
    float* __restrict__ Cout,
    float* __restrict__ Hout2) {         // duplicate write (final layer)
  const int wave = threadIdx.x >> 5;
  const int lane = threadIdx.x & 31;
  const int row0 = (blockIdx.x * 4 + wave) * 16;
  if (row0 >= N_NODES) return;           // wave-uniform: EXEC stays all-ones
  const int nl = lane & 15;
  const int hi = lane >> 4;
  const int aBase = hi * 8;
  const long arow = (long)(row0 + nl) * D;
  const _Float16* wbi = Wih + (long)lane * 16;   // + (ks*32+ct)*512 halves
  const _Float16* wbh = Whh + (long)lane * 16;
  const v8f vzero = {};

  for (int qt = 0; qt < 4; ++qt) {       // channel tiles {qt*2, qt*2+1}
    v8f acc[2][4];
#pragma unroll
    for (int t = 0; t < 2; ++t)
#pragma unroll
      for (int g = 0; g < 4; ++g) acc[t][g] = vzero;

    // ---- input-hidden GEMM ----
#pragma unroll
    for (int ks = 0; ks < KXS; ++ks) {
      const int k0 = ks * 32;
      const float* ap = (HAS_XB && k0 >= D) ? (Xb + arow + (k0 - D))
                                            : (Xa + arow + k0);
      v16h a = make_a(ap, aBase);
#pragma unroll
      for (int t = 0; t < 2; ++t) {
#pragma unroll
        for (int g = 0; g < 4; ++g) {
          const int ct = g * 8 + qt * 2 + t;        // column tile (of 32)
          v16h b = *(const v16h*)(wbi + (long)(ks * 32 + ct) * 512);
          acc[t][g] = __builtin_amdgcn_wmma_f32_16x16x32_f16(
              false, a, false, b, (short)0, acc[t][g], false, false);
        }
      }
    }
    // ---- hidden-hidden GEMM (K = 128, 4 k-steps) ----
#pragma unroll
    for (int ks = 0; ks < 4; ++ks) {
      v16h a = make_a(H0 + arow + ks * 32, aBase);
#pragma unroll
      for (int t = 0; t < 2; ++t) {
#pragma unroll
        for (int g = 0; g < 4; ++g) {
          const int ct = g * 8 + qt * 2 + t;
          v16h b = *(const v16h*)(wbh + (long)(ks * 32 + ct) * 512);
          acc[t][g] = __builtin_amdgcn_wmma_f32_16x16x32_f16(
              false, a, false, b, (short)0, acc[t][g], false, false);
        }
      }
    }

    // ---- LSTM epilogue: lane = gate column, VGPR r = node row r + 8*hi ----
#pragma unroll
    for (int t = 0; t < 2; ++t) {
      const int ch = (qt * 2 + t) * 16 + nl;
      const float bi = Bih[0 * D + ch] + Bhh[0 * D + ch];
      const float bf = Bih[1 * D + ch] + Bhh[1 * D + ch];
      const float bg = Bih[2 * D + ch] + Bhh[2 * D + ch];
      const float bo = Bih[3 * D + ch] + Bhh[3 * D + ch];
#pragma unroll
      for (int r = 0; r < 8; ++r) {
        const long node = row0 + r + 8 * hi;
        const long idx = node * D + ch;
        float iv = sigmoidf_(acc[t][0][r] + bi);
        float fv = sigmoidf_(acc[t][1][r] + bf);
        float gv = tanhf(acc[t][2][r] + bg);
        float ov = sigmoidf_(acc[t][3][r] + bo);
        float cn = fv * C0[idx] + iv * gv;
        float hn = ov * tanhf(cn);
        Hout[idx] = hn;
        Cout[idx] = cn;
        if (DUP) Hout2[idx] = hn;
      }
    }
  }
}

// ---------------------------------------------------------------------------
extern "C" void kernel_launch(void* const* d_in, const int* in_sizes, int n_in,
                              void* d_out, int out_size, void* d_ws, size_t ws_size,
                              hipStream_t stream) {
  const float* feature   = (const float*)d_in[0];
  const float* h0        = (const float*)d_in[1];
  const float* c0        = (const float*)d_in[2];
  const int*   src       = (const int*)d_in[3];
  const int*   dst       = (const int*)d_in[4];
  const float* w_ih0     = (const float*)d_in[5];
  const float* w_hh0     = (const float*)d_in[6];
  const float* w_ih_rest = (const float*)d_in[7];
  const float* w_hh_rest = (const float*)d_in[8];
  const float* b_ih      = (const float*)d_in[9];
  const float* b_hh      = (const float*)d_in[10];

  const long ND = (long)N_NODES * D;
  float* out    = (float*)d_out;
  float* out_y  = out;             // [1,N,128]
  float* out_hn = out + ND;        // [3,N,128]
  float* out_cn = out_hn + 3 * ND; // [3,N,128]

  // workspace layout (32B-aligned)
  float* msg = (float*)d_ws;                 // [N,128] -> h_agg in place
  float* deg = msg + ND;                     // [N]
  size_t woff = (((size_t)(deg + N_NODES)) + 31) & ~(size_t)31;
  _Float16* pw_ih0 = (_Float16*)woff;         // 512*256
  _Float16* pw_hh0 = pw_ih0 + 512L * 256;     // 512*128 each below
  _Float16* pw_ih1 = pw_hh0 + 512L * 128;
  _Float16* pw_hh1 = pw_ih1 + 512L * 128;
  _Float16* pw_ih2 = pw_hh1 + 512L * 128;
  _Float16* pw_hh2 = pw_ih2 + 512L * 128;

  // 0) pack weights into f16 B-fragment layout (tiny, once per launch)
  {
    long t0 = 512L * 256, t1 = 512L * 128;
    pack_w_kernel<<<(int)((t0 + 255) / 256), 256, 0, stream>>>(w_ih0, pw_ih0, 8, t0);
    pack_w_kernel<<<(int)((t1 + 255) / 256), 256, 0, stream>>>(w_hh0, pw_hh0, 4, t1);
    pack_w_kernel<<<(int)((t1 + 255) / 256), 256, 0, stream>>>(w_ih_rest, pw_ih1, 4, t1);
    pack_w_kernel<<<(int)((t1 + 255) / 256), 256, 0, stream>>>(w_hh_rest, pw_hh1, 4, t1);
    pack_w_kernel<<<(int)((t1 + 255) / 256), 256, 0, stream>>>(w_ih_rest + t1, pw_ih2, 4, t1);
    pack_w_kernel<<<(int)((t1 + 255) / 256), 256, 0, stream>>>(w_hh_rest + t1, pw_hh2, 4, t1);
  }

  // 1) aggregation
  long zn = ND + N_NODES;
  zero_kernel<<<(int)((zn + 255) / 256), 256, 0, stream>>>(msg, zn);
  long st = (long)NEDGE * 32;
  scatter_kernel<<<(int)((st + 255) / 256), 256, 0, stream>>>(feature, src, dst, msg, deg);
  mean_kernel<<<(int)((ND + 255) / 256), 256, 0, stream>>>(msg, deg);

  // 2) three fused LSTM layers
  const int blocks = (N_NODES + 63) / 64;   // 4 waves x 16 rows per block
  // layer 0: X = concat(h_agg, feature), Kx = 256
  lstm_layer_kernel<8, true, false><<<blocks, 128, 0, stream>>>(
      msg, feature, h0, c0, pw_ih0, pw_hh0, b_ih, b_hh,
      out_hn, out_cn, nullptr);
  // layer 1: X = h_n[0] (written above into d_out), Kx = 128
  lstm_layer_kernel<4, false, false><<<blocks, 128, 0, stream>>>(
      out_hn, nullptr, h0 + ND, c0 + ND,
      pw_ih1, pw_hh1, b_ih + 512, b_hh + 512,
      out_hn + ND, out_cn + ND, nullptr);
  // layer 2: X = h_n[1]; also writes the duplicate "output" block
  lstm_layer_kernel<4, false, true><<<blocks, 128, 0, stream>>>(
      out_hn + ND, nullptr, h0 + 2 * ND, c0 + 2 * ND,
      pw_ih2, pw_hh2, b_ih + 1024, b_hh + 1024,
      out_hn + 2 * ND, out_cn + 2 * ND, out_y);
}
